// SubgraphCompressorDecompressor_17514876633175
// MI455X (gfx1250) — matrix operand order repaired and checked by reading, not compiled, not run on hardware
//
#include <hip/hip_runtime.h>
#include <hip/hip_bf16.h>
#include <math.h>

#define FDIM 128
#define NBINS 4096

typedef __attribute__((ext_vector_type(16))) __bf16 v16bf;
typedef __attribute__((ext_vector_type(8)))  __bf16 v8bf;
typedef __attribute__((ext_vector_type(8)))  float  v8f;

// ---------------------------------------------------------------------------
// Aggregation kernels (mean over in-neighbors)
// ---------------------------------------------------------------------------
__global__ void deg_kernel(const int* __restrict__ tgt, float* __restrict__ cnt, int E) {
    int e = blockIdx.x * blockDim.x + threadIdx.x;
    if (e < E) atomicAdd(&cnt[tgt[e]], 1.0f);
}

// One thread per (edge, 4-feature chunk): consecutive threads share an edge row.
__global__ void scatter_kernel(const float* __restrict__ X,
                               const int* __restrict__ src, const int* __restrict__ tgt,
                               float* __restrict__ S, int E) {
    long t = (long)blockIdx.x * blockDim.x + threadIdx.x;
    long total = (long)E * (FDIM / 4);
    if (t >= total) return;
    int e  = (int)(t / (FDIM / 4));
    int f4 = (int)(t % (FDIM / 4)) * 4;
    int sn = src[e], dn = tgt[e];
    float4 v = *(const float4*)(X + (long)sn * FDIM + f4);
    float* out = S + (long)dn * FDIM + f4;
    atomicAdd(out + 0, v.x);
    atomicAdd(out + 1, v.y);
    atomicAdd(out + 2, v.z);
    atomicAdd(out + 3, v.w);
}

// mean = sum / max(deg,1), emitted directly as bf16 (means are only GEMM A operands)
__global__ void divide_cvt_kernel(const float* __restrict__ S, const float* __restrict__ cnt,
                                  __bf16* __restrict__ out, long nElem) {
    long t = (long)blockIdx.x * blockDim.x + threadIdx.x;
    if (t < nElem) {
        int row = (int)(t / FDIM);
        out[t] = (__bf16)(S[t] / fmaxf(cnt[row], 1.0f));
    }
}

__global__ void cvt_bf16_kernel(const float* __restrict__ in, __bf16* __restrict__ out, long nElem) {
    long t = (long)blockIdx.x * blockDim.x + threadIdx.x;
    if (t < nElem) out[t] = (__bf16)in[t];
}

// ---------------------------------------------------------------------------
// Weight pre-swizzle: f32 row-major [K=128][N=128]  ->  bf16 B-fragment layout.
// out[((tn*4 + kstep)*32 + lane)*16 + j] = W[kstep*32 + (lane&16 ? 8:0) + (j&7) + (j&8 ? 16:0)][tn*16 + (lane&15)]
// so each lane's 16-element fragment is one contiguous 32-byte load.
// ---------------------------------------------------------------------------
__global__ void swizzle_W_kernel(const float* __restrict__ W, __bf16* __restrict__ out) {
    int t = blockIdx.x * blockDim.x + threadIdx.x;
    if (t >= FDIM * FDIM) return;
    int j     = t & 15;
    int lane  = (t >> 4) & 31;
    int kstep = (t >> 9) & 3;
    int tn    = (t >> 11);
    int krow  = kstep * 32 + ((lane & 16) ? 8 : 0) + (j & 7) + ((j & 8) ? 16 : 0);
    int col   = tn * 16 + (lane & 15);
    out[t] = (__bf16)W[krow * FDIM + col];
}

// ---------------------------------------------------------------------------
// WMMA fragment loads (all-bf16, no inner-loop conversion)
// ---------------------------------------------------------------------------
__device__ inline v16bf load_A_bf(const __bf16* __restrict__ rowBase, int k0, int lane) {
    // A 16x32 bf16: lanes<16 -> K=k0..k0+7 & k0+16..k0+23 ; lanes>=16 -> +8
    int off = k0 + ((lane & 16) ? 8 : 0);
    v8bf lo = *(const v8bf*)(rowBase + off);
    v8bf hi = *(const v8bf*)(rowBase + off + 16);
    return __builtin_shufflevector(lo, hi, 0, 1, 2, 3, 4, 5, 6, 7,
                                   8, 9, 10, 11, 12, 13, 14, 15);
}

__device__ inline v16bf load_B_swz(const __bf16* __restrict__ Wswz, int tnT, int kidx, int lane) {
    return *(const v16bf*)(Wswz + (((tnT * 4 + kidx) * 32 + lane) << 4));
}

// h = X @ Wself + M @ Wneigh + bias     (SAGE layer, fused dual GEMM)
__global__ void gemm_dual_kernel(const __bf16* __restrict__ Xb, const __bf16* __restrict__ Mb,
                                 const __bf16* __restrict__ WselfS, const __bf16* __restrict__ WneighS,
                                 const float* __restrict__ bias, float* __restrict__ H, int Nrows) {
    int wave = threadIdx.x >> 5;
    int lane = threadIdx.x & 31;
    int tile = blockIdx.x * (blockDim.x >> 5) + wave;
    const int tilesN = FDIM / 16;                    // 8
    int tilesM = (Nrows + 15) / 16;
    if (tile >= tilesM * tilesN) return;
    int tm  = tile / tilesN;
    int tnT = tile % tilesN;
    int row = tm * 16 + (lane & 15);
    int lrow = row < Nrows ? row : (Nrows - 1);
    const __bf16* xr = Xb + (long)lrow * FDIM;
    const __bf16* mr = Mb + (long)lrow * FDIM;

    v8f c = {};
#pragma unroll
    for (int kidx = 0; kidx < 4; ++kidx) {
        int k0 = kidx * 32;
        v16bf a0 = load_A_bf(xr, k0, lane);
        v16bf b0 = load_B_swz(WselfS, tnT, kidx, lane);
        c = __builtin_amdgcn_wmma_f32_16x16x32_bf16(false, a0, false, b0, (short)0, c, false, false);
        v16bf a1 = load_A_bf(mr, k0, lane);
        v16bf b1 = load_B_swz(WneighS, tnT, kidx, lane);
        c = __builtin_amdgcn_wmma_f32_16x16x32_bf16(false, a1, false, b1, (short)0, c, false, false);
    }
    int col   = tnT * 16 + (lane & 15);
    int rbase = tm * 16 + ((lane & 16) ? 8 : 0);
    float bv  = bias[col];
#pragma unroll
    for (int r = 0; r < 8; ++r) {
        int rr = rbase + r;
        if (rr < Nrows) H[(long)rr * FDIM + col] = c[r] + bv;
    }
}

// Out = Agg @ W + bias, with seed override for flagged rows; mode 1 applies ReLU.
__global__ void gemm_diffuse_kernel(const __bf16* __restrict__ Mb, const __bf16* __restrict__ WS,
                                    const float* __restrict__ bias, const int* __restrict__ flag,
                                    const float* __restrict__ seed, float* __restrict__ Out,
                                    int Nrows, int mode) {
    int wave = threadIdx.x >> 5;
    int lane = threadIdx.x & 31;
    int tile = blockIdx.x * (blockDim.x >> 5) + wave;
    const int tilesN = FDIM / 16;
    int tilesM = (Nrows + 15) / 16;
    if (tile >= tilesM * tilesN) return;
    int tm  = tile / tilesN;
    int tnT = tile % tilesN;
    int row = tm * 16 + (lane & 15);
    int lrow = row < Nrows ? row : (Nrows - 1);
    const __bf16* ar = Mb + (long)lrow * FDIM;

    v8f c = {};
#pragma unroll
    for (int kidx = 0; kidx < 4; ++kidx) {
        v16bf a = load_A_bf(ar, kidx * 32, lane);
        v16bf b = load_B_swz(WS, tnT, kidx, lane);
        c = __builtin_amdgcn_wmma_f32_16x16x32_bf16(false, a, false, b, (short)0, c, false, false);
    }
    int col   = tnT * 16 + (lane & 15);
    int rbase = tm * 16 + ((lane & 16) ? 8 : 0);
    float bv  = bias[col];
#pragma unroll
    for (int r = 0; r < 8; ++r) {
        int rr = rbase + r;
        if (rr < Nrows) {
            float v = c[r] + bv;
            if (flag[rr]) v = seed[(long)rr * FDIM + col];
            if (mode == 1) v = fmaxf(v, 0.0f);
            Out[(long)rr * FDIM + col] = v;
        }
    }
}

// ---------------------------------------------------------------------------
// Scorer + deterministic top-K via histogram threshold
// ---------------------------------------------------------------------------
__global__ void score_kernel(const float* __restrict__ H, const float* __restrict__ w,
                             const float* __restrict__ b, float* __restrict__ score, int Nrows) {
    int node = blockIdx.x * (blockDim.x >> 5) + (threadIdx.x >> 5);
    int lane = threadIdx.x & 31;
    if (node >= Nrows) return;
    float4 hv = *(const float4*)(H + (long)node * FDIM + lane * 4);
    float4 wv = *(const float4*)(w + lane * 4);
    float acc = hv.x * wv.x + hv.y * wv.y + hv.z * wv.z + hv.w * wv.w;
#pragma unroll
    for (int off = 16; off > 0; off >>= 1) acc += __shfl_xor(acc, off, 32);
    if (lane == 0) score[node] = 1.0f / (1.0f + expf(-(acc + b[0])));
}

__global__ void hist_kernel(const float* __restrict__ score, int* __restrict__ bins, int Nrows) {
    int i = blockIdx.x * blockDim.x + threadIdx.x;
    if (i < Nrows) {
        int b = (int)(score[i] * (float)NBINS);
        b = b < 0 ? 0 : (b >= NBINS ? NBINS - 1 : b);
        atomicAdd(&bins[b], 1);
    }
}

__global__ void thresh_kernel(const int* __restrict__ bins, int* __restrict__ thr, int K) {
    if (threadIdx.x == 0 && blockIdx.x == 0) {
        int acc = 0;
        for (int b = NBINS - 1; b >= 0; --b) {
            if (acc + bins[b] >= K) { thr[0] = b; thr[1] = K - acc; return; }
            acc += bins[b];
        }
        thr[0] = 0; thr[1] = K - acc;
    }
}

__global__ void block_count_kernel(const float* __restrict__ score, const int* __restrict__ thr,
                                   int* __restrict__ blockCounts, int Nrows) {
    __shared__ int scnt;
    if (threadIdx.x == 0) scnt = 0;
    __syncthreads();
    int i = blockIdx.x * blockDim.x + threadIdx.x;
    int pred = 0;
    if (i < Nrows) {
        int b = (int)(score[i] * (float)NBINS);
        b = b < 0 ? 0 : (b >= NBINS ? NBINS - 1 : b);
        pred = (b == thr[0]);
    }
    unsigned long long m = __ballot(pred);
    if ((threadIdx.x & 31) == 0) atomicAdd(&scnt, __popcll(m));
    __syncthreads();
    if (threadIdx.x == 0) blockCounts[blockIdx.x] = scnt;
}

__global__ void scan_kernel(const int* __restrict__ blockCounts, int* __restrict__ blockOffsets, int nb) {
    if (threadIdx.x == 0 && blockIdx.x == 0) {
        int acc = 0;
        for (int i = 0; i < nb; ++i) { blockOffsets[i] = acc; acc += blockCounts[i]; }
    }
}

__global__ void select_kernel(const float* __restrict__ score, const int* __restrict__ thr,
                              const int* __restrict__ blockOffsets, int* __restrict__ flag, int Nrows) {
    __shared__ int waveCnt[8];
    __shared__ int waveOff[8];
    int i    = blockIdx.x * blockDim.x + threadIdx.x;
    int lane = threadIdx.x & 31;
    int wave = threadIdx.x >> 5;
    int t = thr[0], rem = thr[1];
    int bin = -1;
    if (i < Nrows) {
        int b = (int)(score[i] * (float)NBINS);
        bin = b < 0 ? 0 : (b >= NBINS ? NBINS - 1 : b);
    }
    int pred = (bin == t);
    unsigned long long m = __ballot(pred);
    int before = __popcll(m & (((unsigned long long)1 << lane) - 1ull));
    if (lane == 0) waveCnt[wave] = __popcll(m);
    __syncthreads();
    if (threadIdx.x == 0) {
        int acc = 0;
        for (int w = 0; w < 8; ++w) { waveOff[w] = acc; acc += waveCnt[w]; }
    }
    __syncthreads();
    if (i < Nrows) {
        int f = 0;
        if (bin > t) f = 1;
        else if (pred) {
            int rank = blockOffsets[blockIdx.x] + waveOff[wave] + before;
            f = (rank < rem) ? 1 : 0;
        }
        flag[i] = f;
    }
}

// z[i] = flag[i] ? h[i] * score[i] : 0
__global__ void gate_kernel(const float* __restrict__ H, const float* __restrict__ score,
                            const int* __restrict__ flag, float* __restrict__ Z, long nElem) {
    long t = (long)blockIdx.x * blockDim.x + threadIdx.x;
    if (t < nElem) {
        int row = (int)(t / FDIM);
        Z[t] = flag[row] ? H[t] * score[row] : 0.0f;
    }
}

// ---------------------------------------------------------------------------
// Launch
// ---------------------------------------------------------------------------
static inline size_t align256(size_t x) { return (x + 255) & ~(size_t)255; }

extern "C" void kernel_launch(void* const* d_in, const int* in_sizes, int n_in,
                              void* d_out, int out_size, void* d_ws, size_t ws_size,
                              hipStream_t stream) {
    const float* x      = (const float*)d_in[0];
    const int*   src    = (const int*)d_in[1];
    const int*   tgt    = (const int*)d_in[2];
    const float* Wself  = (const float*)d_in[3];
    const float* Wneigh = (const float*)d_in[4];
    const float* bpack  = (const float*)d_in[5];
    const float* wscore = (const float*)d_in[6];
    const float* bscore = (const float*)d_in[7];
    const float* Wu1    = (const float*)d_in[8];
    const float* bu1    = (const float*)d_in[9];
    const float* Wu2    = (const float*)d_in[10];
    const float* bu2    = (const float*)d_in[11];

    const int N = in_sizes[0] / FDIM;
    const int E = in_sizes[1];
    const int K = (N / 4) > 1 ? (N / 4) : 1;
    const long NF = (long)N * FDIM;

    // workspace carve-up
    char* ws = (char*)d_ws;
    size_t off = 0;
    float*  cnt    = (float*)(ws + off);  off = align256(off + (size_t)N * 4);
    float*  aggbuf = (float*)(ws + off);  off = align256(off + (size_t)NF * 4);
    float*  h      = (float*)(ws + off);  off = align256(off + (size_t)NF * 4);   // reused as d
    float*  z      = (float*)(ws + off);  off = align256(off + (size_t)NF * 4);
    __bf16* xb     = (__bf16*)(ws + off); off = align256(off + (size_t)NF * 2);
    __bf16* mb     = (__bf16*)(ws + off); off = align256(off + (size_t)NF * 2);
    __bf16* WselfS = (__bf16*)(ws + off); off = align256(off + (size_t)FDIM * FDIM * 2);
    __bf16* WneighS= (__bf16*)(ws + off); off = align256(off + (size_t)FDIM * FDIM * 2);
    __bf16* Wu1S   = (__bf16*)(ws + off); off = align256(off + (size_t)FDIM * FDIM * 2);
    __bf16* Wu2S   = (__bf16*)(ws + off); off = align256(off + (size_t)FDIM * FDIM * 2);
    float*  score  = (float*)(ws + off);  off = align256(off + (size_t)N * 4);
    int*    bins   = (int*)(ws + off);    off = align256(off + (size_t)NBINS * 4);
    int*    thr    = (int*)(ws + off);    off = align256(off + 2 * 4);
    const int nb   = (N + 255) / 256;
    int*    bcnt   = (int*)(ws + off);    off = align256(off + (size_t)nb * 4);
    int*    boff   = (int*)(ws + off);    off = align256(off + (size_t)nb * 4);
    int*    flag   = (int*)(ws + off);    off = align256(off + (size_t)N * 4);
    float*  out    = (float*)d_out;

    const int TB = 256;
    const long scatterT = (long)E * (FDIM / 4);
    const int  scatterB = (int)((scatterT + TB - 1) / TB);
    const int  elemB    = (int)((NF + TB - 1) / TB);
    const int  nodeB    = nb;
    const int  tiles    = ((N + 15) / 16) * (FDIM / 16);
    const int  gemmB    = (tiles + 7) / 8;        // 8 waves / block
    const int  swzB     = (FDIM * FDIM + TB - 1) / TB;

    // ---- one-time prep per call: degrees, weight swizzles, x -> bf16 ----
    hipMemsetAsync(cnt, 0, (size_t)N * 4, stream);
    hipMemsetAsync(bins, 0, (size_t)NBINS * 4, stream);
    deg_kernel<<<(E + TB - 1) / TB, TB, 0, stream>>>(tgt, cnt, E);
    swizzle_W_kernel<<<swzB, TB, 0, stream>>>(Wself,  WselfS);
    swizzle_W_kernel<<<swzB, TB, 0, stream>>>(Wneigh, WneighS);
    swizzle_W_kernel<<<swzB, TB, 0, stream>>>(Wu1,    Wu1S);
    swizzle_W_kernel<<<swzB, TB, 0, stream>>>(Wu2,    Wu2S);
    cvt_bf16_kernel<<<elemB, TB, 0, stream>>>(x, xb, NF);

    // ---- compress: h = x@Wself + mean_agg(x)@Wneigh + b ----
    hipMemsetAsync(aggbuf, 0, (size_t)NF * 4, stream);
    scatter_kernel<<<scatterB, TB, 0, stream>>>(x, src, tgt, aggbuf, E);
    divide_cvt_kernel<<<elemB, TB, 0, stream>>>(aggbuf, cnt, mb, NF);
    gemm_dual_kernel<<<gemmB, TB, 0, stream>>>(xb, mb, WselfS, WneighS, bpack, h, N);

    // scorer + deterministic top-K flags
    score_kernel<<<(N * 32 + TB - 1) / TB, TB, 0, stream>>>(h, wscore, bscore, score, N);
    hist_kernel<<<nodeB, TB, 0, stream>>>(score, bins, N);
    thresh_kernel<<<1, 32, 0, stream>>>(bins, thr, K);
    block_count_kernel<<<nodeB, TB, 0, stream>>>(score, thr, bcnt, N);
    scan_kernel<<<1, 32, 0, stream>>>(bcnt, boff, nb);
    select_kernel<<<nodeB, TB, 0, stream>>>(score, thr, boff, flag, N);
    gate_kernel<<<elemB, TB, 0, stream>>>(h, score, flag, z, NF);

    // ---- decompress layer 1: d = relu(seed-override(mean_agg(z)@Wu1 + b1)) ----
    hipMemsetAsync(aggbuf, 0, (size_t)NF * 4, stream);
    scatter_kernel<<<scatterB, TB, 0, stream>>>(z, src, tgt, aggbuf, E);
    divide_cvt_kernel<<<elemB, TB, 0, stream>>>(aggbuf, cnt, mb, NF);
    float* d = h;   // h no longer needed: reuse buffer for d
    gemm_diffuse_kernel<<<gemmB, TB, 0, stream>>>(mb, Wu1S, bu1, flag, z, d, N, /*relu*/1);

    // ---- decompress layer 2: out = seed-override(mean_agg(d)@Wu2 + b2) ----
    hipMemsetAsync(aggbuf, 0, (size_t)NF * 4, stream);
    scatter_kernel<<<scatterB, TB, 0, stream>>>(d, src, tgt, aggbuf, E);
    divide_cvt_kernel<<<elemB, TB, 0, stream>>>(aggbuf, cnt, mb, NF);
    gemm_diffuse_kernel<<<gemmB, TB, 0, stream>>>(mb, Wu2S, bu2, flag, d, out, N, /*no relu*/0);
}
